// StaticProjector_36163624632622
// MI455X (gfx1250) — compile-verified, matrix operand order checked by reading, not run on hardware
//
#include <hip/hip_runtime.h>
#include <hip/hip_bf16.h>

// Problem constants (from reference)
#define B_ 16
#define CT_ 512
#define C_ 256
#define L_ 512
#define N_ 4096   // H*W
#define WAVES_ 8  // waves (n-tiles) per workgroup in the fused attention kernel

typedef __attribute__((ext_vector_type(16))) _Float16 v16h;
typedef __attribute__((ext_vector_type(8)))  float    v8f;

#define DEV static __device__ __forceinline__

// WMMA 16x16x32 f16 fragment element -> K index mapping (CDNA5 ISA 7.12.2):
// lanes 0-15 hold K {0..7, 16..23}, lanes 16-31 hold K {8..15, 24..31};
// element e (0..15): k = 8*half + (e<8 ? e : e+8)

// Load a 32-K f16 fragment whose lane-row/col data is contiguous at `base`.
// Works for global or LDS pointers (compiler infers address space).
DEV v16h ldfrag_h(const _Float16* base, int half) {
  union { uint4 u[2]; v16h v; } r;
  r.u[0] = *(const uint4*)(base + 8 * half);        // K = 8h .. 8h+7
  r.u[1] = *(const uint4*)(base + 16 + 8 * half);   // K = 16+8h .. 16+8h+7
  return r.v;
}

// Build fragment from contiguous f32 data (convert to f16).
DEV v16h cvtfrag_f32(const float* base, int half) {
  v16h f;
#pragma unroll
  for (int e = 0; e < 16; ++e) {
    int k = 8 * half + (e < 8 ? e : e + 8);
    f[e] = (_Float16)base[k];
  }
  return f;
}

// Build fragment from strided f32 data (gather along K; coalesced across lanes).
DEV v16h cvtfrag_f32_s(const float* base, int half, int stride) {
  v16h f;
#pragma unroll
  for (int e = 0; e < 16; ++e) {
    int k = 8 * half + (e < 8 ? e : e + 8);
    f[e] = (_Float16)base[(size_t)k * stride];
  }
  return f;
}

DEV v8f wmma_f16(v16h a, v16h b, v8f c) {
  return __builtin_amdgcn_wmma_f32_16x16x32_f16(false, a, false, b, (short)0, c,
                                                false, false);
}

// CDNA5 async global->LDS copy of 16 bytes (per lane), tracked by ASYNCcnt.
DEV void async_copy_b128(unsigned lds_off, const _Float16* gaddr) {
  asm volatile("global_load_async_to_lds_b128 %0, %1, off"
               :: "v"(lds_off), "v"(gaddr) : "memory");
}
DEV void async_wait0() {
  asm volatile("s_wait_asynccnt 0" ::: "memory");
}

// ---------------------------------------------------------------------------
// Kernel 1: k/v projections from tokens (f32) into f16 workspace.
//   mode 0: kT[b][l][c] = sum_ct Wk[c][ct]*tokens[b][ct][l] + bk[c]   (L x C)
//   mode 1: v [b][c][l] = sum_ct Wv[c][ct]*tokens[b][ct][l] + bv[c]   (C x L)
// ---------------------------------------------------------------------------
__global__ __launch_bounds__(32, 1)
void kv_proj_kernel(const float* __restrict__ tokens,
                    const float* __restrict__ Wk, const float* __restrict__ bk,
                    const float* __restrict__ Wv, const float* __restrict__ bv,
                    _Float16* __restrict__ kT, _Float16* __restrict__ v) {
  const int lane = threadIdx.x;
  const int tid  = lane & 15;
  const int half = lane >> 4;
  const int b    = blockIdx.y;
  const float* tok = tokens + (size_t)b * CT_ * L_;

  v8f acc = {};
  if (blockIdx.z == 0) {
    const int lt = blockIdx.x >> 4, ctile = blockIdx.x & 15;
    const int l0 = lt * 16, c0 = ctile * 16;
#pragma unroll
    for (int kc = 0; kc < CT_ / 32; ++kc) {
      v16h a  = cvtfrag_f32_s(tok + (size_t)(kc * 32) * L_ + l0 + tid, half, L_);
      v16h bb = cvtfrag_f32(Wk + (size_t)(c0 + tid) * CT_ + kc * 32, half);
      acc = wmma_f16(a, bb, acc);
    }
    const float bias = bk[c0 + tid];
    _Float16* dst = kT + (size_t)b * L_ * C_;
#pragma unroll
    for (int r = 0; r < 8; ++r)
      dst[(size_t)(l0 + r + 8 * half) * C_ + c0 + tid] = (_Float16)(acc[r] + bias);
  } else {
    const int ctile = blockIdx.x >> 5, lt = blockIdx.x & 31;
    const int c0 = ctile * 16, l0 = lt * 16;
#pragma unroll
    for (int kc = 0; kc < CT_ / 32; ++kc) {
      v16h a  = cvtfrag_f32(Wv + (size_t)(c0 + tid) * CT_ + kc * 32, half);
      v16h bb = cvtfrag_f32_s(tok + (size_t)(kc * 32) * L_ + l0 + tid, half, L_);
      acc = wmma_f16(a, bb, acc);
    }
    _Float16* dst = v + (size_t)b * C_ * L_;
#pragma unroll
    for (int r = 0; r < 8; ++r) {
      const float bias = bv[c0 + r + 8 * half];
      dst[(size_t)(c0 + r + 8 * half) * L_ + l0 + tid] = (_Float16)(acc[r] + bias);
    }
  }
}

// ---------------------------------------------------------------------------
// Kernel 2: q projection: q[b][n][c] = sum_c' Wq[c][c']*feature[b][c'][n] + bq[c]
// ---------------------------------------------------------------------------
__global__ __launch_bounds__(32, 1)
void q_proj_kernel(const float* __restrict__ feature,
                   const float* __restrict__ Wq, const float* __restrict__ bq,
                   _Float16* __restrict__ q) {
  const int lane = threadIdx.x;
  const int tid  = lane & 15;
  const int half = lane >> 4;
  const int b    = blockIdx.y;
  const int nt   = blockIdx.x >> 4, ctile = blockIdx.x & 15;
  const int n0 = nt * 16, c0 = ctile * 16;
  const float* fb = feature + (size_t)b * C_ * N_;

  v8f acc = {};
#pragma unroll
  for (int kc = 0; kc < C_ / 32; ++kc) {
    v16h a  = cvtfrag_f32_s(fb + (size_t)(kc * 32) * N_ + n0 + tid, half, N_);
    v16h bb = cvtfrag_f32(Wq + (size_t)(c0 + tid) * C_ + kc * 32, half);
    acc = wmma_f16(a, bb, acc);
  }
  const float bias = bq[c0 + tid];
  _Float16* dst = q + (size_t)b * N_ * C_;
#pragma unroll
  for (int r = 0; r < 8; ++r)
    dst[(size_t)(n0 + r + 8 * half) * C_ + c0 + tid] = (_Float16)(acc[r] + bias);
}

// ---------------------------------------------------------------------------
// Kernel 3: fused attention, 8 waves (8 n-tiles) per workgroup.
// The FULL per-batch kT matrix (512x256 f16, bank-padded: 264 KB) is staged
// into LDS once per WG with async global->LDS B128 copies; both softmax
// passes read it from LDS barrier-free. The same LDS region is then reused
// to hold the FULL v matrix (256x512 f16, padded) for the kqv phase.
// -> kT+v+q global traffic: 576 KB per WG (~0.3 GB total), 3 barriers per WG.
//  S^T[l][n] = sum_c kT[l][c]*q[n][c] ; softmax over l (lane-local + 1 shfl)
//  P^T accumulator layout == B-fragment layout for out^T = v * P^T.
// ---------------------------------------------------------------------------
#define KROW_ 264   // 256+8 pad halfs: 132 dwords/row -> every-bank-exactly-2x
#define VROW_ 520   // 512+8 pad halfs: 260 dwords/row -> every-bank-exactly-2x
#define SMEM_BYTES_ (L_ * KROW_ * 2)   // 270336 B; v view (256*520*2) fits too

__global__ __launch_bounds__(32 * WAVES_, 1)
void attn_kernel(const _Float16* __restrict__ kT, const _Float16* __restrict__ vv,
                 const _Float16* __restrict__ q, const float* __restrict__ feature,
                 float* __restrict__ out) {
  extern __shared__ _Float16 smem[];   // 264 KB, dynamically allocated

  const int flat = threadIdx.x;
  const int lane = flat & 31;
  const int wave = flat >> 5;
  const int tid  = lane & 15;
  const int half = lane >> 4;
  const int b    = blockIdx.y;
  const int n0   = (blockIdx.x * WAVES_ + wave) * 16;

  const _Float16* kTb = kT + (size_t)b * L_ * C_;
  const _Float16* vb  = vv + (size_t)b * C_ * L_;
  const _Float16* qb  = q  + (size_t)b * N_ * C_;
  const unsigned smem_base = (unsigned)(size_t)(void*)smem;

  // ---- Stage ALL of kT: 512 rows x 32 groups-of-8-halfs = 16384 B128s ----
  {
#pragma unroll 8
    for (int g = 0; g < 64; ++g) {
      const int grp = flat + g * 256;          // 0..16383
      const int row = grp >> 5, colg = grp & 31;
      async_copy_b128(smem_base + (unsigned)((row * KROW_ + colg * 8) * 2),
                      kTb + (size_t)row * C_ + colg * 8);
    }
    async_wait0();
    __syncthreads();
  }

  // q^T B-fragments: lane = n column, K = c (contiguous per lane). Resident.
  v16h qB[C_ / 32];
#pragma unroll
  for (int kc = 0; kc < C_ / 32; ++kc)
    qB[kc] = ldfrag_h(qb + (size_t)(n0 + tid) * C_ + kc * 32, half);

  // ---- Pass 1: row max over L (kT read from LDS, no barriers) ----
  float m = -3.0e38f;
  for (int t = 0; t < L_ / 16; ++t) {
    v8f acc = {};
#pragma unroll
    for (int kc = 0; kc < C_ / 32; ++kc) {
      v16h a = ldfrag_h(smem + (t * 16 + tid) * KROW_ + kc * 32, half);
      acc = wmma_f16(a, qB[kc], acc);
    }
#pragma unroll
    for (int r = 0; r < 8; ++r) m = fmaxf(m, acc[r]);
  }
  m = fmaxf(m, __shfl_xor(m, 16));

  // ---- Pass 2: recompute scores, exp, pack P^T as f16 B-fragments ----
  v16h P[L_ / 32];
  float s = 0.0f;
#pragma unroll
  for (int t = 0; t < L_ / 16; ++t) {
    v8f acc = {};
#pragma unroll
    for (int kc = 0; kc < C_ / 32; ++kc) {
      v16h a = ldfrag_h(smem + (t * 16 + tid) * KROW_ + kc * 32, half);
      acc = wmma_f16(a, qB[kc], acc);
    }
#pragma unroll
    for (int r = 0; r < 8; ++r) {
      const float e = __expf(acc[r] - m);
      s += e;
      P[t >> 1][(t & 1) * 8 + r] = (_Float16)e;  // C-layout row == B-frag K slot
    }
  }
  s += __shfl_xor(s, 16);
  const float inv = 1.0f / s;

  // ---- Swap LDS contents: kT done, stage ALL of v into the same region ----
  __syncthreads();   // everyone done reading kT
  {
#pragma unroll 8
    for (int g = 0; g < 64; ++g) {
      const int grp = flat + g * 256;          // 0..16383
      const int row = grp >> 6, colg = grp & 63;
      async_copy_b128(smem_base + (unsigned)((row * VROW_ + colg * 8) * 2),
                      vb + (size_t)row * L_ + colg * 8);
    }
    async_wait0();
    __syncthreads();
  }

  // ---- Pass 3: out^T tiles (c x n), K = L in chunks of 32, v from LDS ----
  v8f o[C_ / 16];
#pragma unroll
  for (int ct = 0; ct < C_ / 16; ++ct) { v8f z = {}; o[ct] = z; }
#pragma unroll
  for (int kb = 0; kb < L_ / 32; ++kb) {
#pragma unroll
    for (int ct = 0; ct < C_ / 16; ++ct) {
      v16h a = ldfrag_h(smem + (ct * 16 + tid) * VROW_ + kb * 32, half);
      o[ct] = wmma_f16(a, P[kb], o[ct]);
    }
  }

  // Residual add + store: lane = n (contiguous f32 across lanes), reg = c row.
  const float* fb = feature + (size_t)b * C_ * N_;
  float* ob = out + (size_t)b * C_ * N_;
#pragma unroll
  for (int ct = 0; ct < C_ / 16; ++ct) {
#pragma unroll
    for (int r = 0; r < 8; ++r) {
      const size_t idx = (size_t)(ct * 16 + r + 8 * half) * N_ + n0 + tid;
      ob[idx] = fb[idx] + inv * o[ct][r];
    }
  }
}

// ---------------------------------------------------------------------------
extern "C" void kernel_launch(void* const* d_in, const int* in_sizes, int n_in,
                              void* d_out, int out_size, void* d_ws, size_t ws_size,
                              hipStream_t stream) {
  const float* tokens  = (const float*)d_in[0];
  const float* feature = (const float*)d_in[1];
  const float* Wk      = (const float*)d_in[2];
  const float* bk      = (const float*)d_in[3];
  const float* Wq      = (const float*)d_in[4];
  const float* bq      = (const float*)d_in[5];
  const float* Wv      = (const float*)d_in[6];
  const float* bv      = (const float*)d_in[7];
  float* out = (float*)d_out;

  // Workspace: kT (B,L,C) f16 | v (B,C,L) f16 | q (B,N,C) f16  => ~40 MB
  _Float16* kT = (_Float16*)d_ws;
  _Float16* v  = kT + (size_t)B_ * L_ * C_;
  _Float16* q  = v  + (size_t)B_ * C_ * L_;

  kv_proj_kernel<<<dim3(512, B_, 2), 32, 0, stream>>>(tokens, Wk, bk, Wv, bv, kT, v);
  q_proj_kernel<<<dim3(4096, B_), 32, 0, stream>>>(feature, Wq, bq, q);
  attn_kernel<<<dim3(N_ / 16 / WAVES_, B_), 32 * WAVES_, SMEM_BYTES_, stream>>>(
      kT, v, q, feature, out);
}
